// GCN_54494545051939
// MI455X (gfx1250) — matrix operand order repaired and checked by reading, not compiled, not run on hardware
//
#include <hip/hip_runtime.h>
#include <hip/hip_bf16.h>

// ---------------------------------------------------------------------------
// GCN forward on gfx1250 (MI455X), wave32 + WMMA bf16.
//   prep:  pack W1/W2/Wfc(f32) -> fragment-ordered bf16 (Wfc padded 11->16)
//   layer: hs = (A @ W) * dinv[row]            (WMMA GEMM, fused row scale)
//          acc[dst] += hs[src]  over edges     (float2 gather + f32 atomics, L2-resident)
//          h = relu(dinv*(acc + hs) + b)       (self-loop folded in as +hs)
//   fc:    out = h @ Wfc + bfc                 (WMMA, tail-masked 11-col store)
// ---------------------------------------------------------------------------

typedef __attribute__((ext_vector_type(16))) __bf16 v16bf;
typedef __attribute__((ext_vector_type(8)))  float  v8f;

union BFrag { uint4 q[2]; v16bf v; };

static __device__ __forceinline__ unsigned short f2bf_bits(float f) {
    union { float f; unsigned int u; } in; in.f = f;
    unsigned int u = in.u + 0x7FFFu + ((in.u >> 16) & 1u);   // round-to-nearest-even
    return (unsigned short)(u >> 16);
}
static __device__ __forceinline__ __bf16 f2bf(float f) {
    union { unsigned short s; __bf16 b; } o; o.s = f2bf_bits(f); return o.b;
}

// ---- weight packing: f32 [K x Wcols] -> fragment-ordered bf16 -------------
// Linear layout: ((kt*ntiles + t)*32 + lane)*16 + e, matching the GEMM's two
// uint4 loads per lane. e -> k = kt*32 + 8*(lane>>4) + (e&7) + 16*(e>>3) ; n = (lane&15)+16t.

__global__ void gcn_pack_w(const float* __restrict__ W, unsigned short* __restrict__ out,
                           int ktiles, int ntiles, int Wcols) {
    int i = blockIdx.x * blockDim.x + threadIdx.x;
    if (i >= ktiles * ntiles * 512) return;
    int e    = i & 15;
    int lane = (i >> 4) & 31;
    int tt   = i >> 9;               // kt*ntiles + t
    int t    = tt % ntiles;
    int kt   = tt / ntiles;
    int k    = (kt << 5) + ((lane >> 4) << 3) + (e & 7) + ((e >> 3) << 4);
    int n    = (lane & 15) + (t << 4);
    float v  = (n < Wcols) ? W[k * Wcols + n] : 0.0f;
    out[i] = f2bf_bits(v);
}

// ---- degree / dinv --------------------------------------------------------

__global__ void gcn_init_deg(float* __restrict__ deg, int n) {
    int i = blockIdx.x * blockDim.x + threadIdx.x;
    if (i < n) deg[i] = 1.0f;                       // self-loop contributes 1
}

__global__ void gcn_degree(const long long* __restrict__ dst, float* __restrict__ deg, int e) {
    int i = blockIdx.x * blockDim.x + threadIdx.x;
    if (i < e) unsafeAtomicAdd(&deg[(int)dst[i]], 1.0f);
}

__global__ void gcn_dinv(const float* __restrict__ deg, float* __restrict__ dinv, int n) {
    int i = blockIdx.x * blockDim.x + threadIdx.x;
    if (i < n) dinv[i] = rsqrtf(deg[i]);            // deg >= 1 always
}

// ---- WMMA GEMM: out[M x outcols] = A[M x K] @ Wp --------------------------
// One wave -> 16 rows x NTILES*16 cols. K multiple of 32, M multiple of 16.
// A fragment: per ISA layout, lane (row = lane&15) needs two contiguous
// 8-float runs: [k0+8*half, +7] and [k0+16+8*half, +7] -> 4x float4 loads.
// B fragments: all NTILES*2 b128 loads issued before a sched_barrier so they
// stay in one clause with maximal MLP; WMMAs then consume with partial waits.

template <int NTILES, bool TAIL>
__global__ void gcn_gemm(const float* __restrict__ A, const unsigned short* __restrict__ Wp,
                         const float* __restrict__ rowscale, const float* __restrict__ bias,
                         float* __restrict__ out, int M, int K, int outld, int outcols) {
    const int lane = threadIdx.x & 31;
    const int wave = threadIdx.x >> 5;
    const int tile = blockIdx.x * (blockDim.x >> 5) + wave;
    if (tile >= (M >> 4)) return;                   // wave-uniform exit (EXEC all-1s)
    const int row0 = tile << 4;
    const int half = lane >> 4;                     // 0: lanes 0-15, 1: lanes 16-31
    const int l15  = lane & 15;

    v8f c[NTILES];
    #pragma unroll
    for (int t = 0; t < NTILES; ++t)
        #pragma unroll
        for (int r = 0; r < 8; ++r) c[t][r] = 0.0f;

    const float* rowp = A + (size_t)(row0 + l15) * K + (half << 3);
    const uint4* wq   = (const uint4*)Wp + lane * 2;

    for (int k0 = 0; k0 < K; k0 += 32) {
        // Issue all loads first: 4x b128 (A rows) + NTILES*2x b128 (B frags).
        const float4* p = (const float4*)(rowp + k0);
        float4 q0 = p[0], q1 = p[1], q2 = p[4], q3 = p[5];

        BFrag b[NTILES];
        const uint4* wk = wq + ((k0 >> 5) * NTILES) * 64;
        #pragma unroll
        for (int t = 0; t < NTILES; ++t) {
            b[t].q[0] = wk[t * 64 + 0];
            b[t].q[1] = wk[t * 64 + 1];
        }

        // A f32 -> bf16 conversion overlaps the outstanding B loads.
        v16bf a;
        a[0]  = f2bf(q0.x); a[1]  = f2bf(q0.y); a[2]  = f2bf(q0.z); a[3]  = f2bf(q0.w);
        a[4]  = f2bf(q1.x); a[5]  = f2bf(q1.y); a[6]  = f2bf(q1.z); a[7]  = f2bf(q1.w);
        a[8]  = f2bf(q2.x); a[9]  = f2bf(q2.y); a[10] = f2bf(q2.z); a[11] = f2bf(q2.w);
        a[12] = f2bf(q3.x); a[13] = f2bf(q3.y); a[14] = f2bf(q3.z); a[15] = f2bf(q3.w);

        // Keep all loads above the WMMA chain: forces distinct B registers,
        // one load clause, and partial loadcnt waits instead of load->wait->wmma.
        __builtin_amdgcn_sched_barrier(0);

        #pragma unroll
        for (int t = 0; t < NTILES; ++t)
            c[t] = __builtin_amdgcn_wmma_f32_16x16x32_bf16(
                       false, a, false, b[t].v, (short)0, c[t], false, false);
    }

    // Epilogue. C/D: VGPR r -> row r (lanes 0-15) / r+8 (lanes 16-31), col = lane&15.
    #pragma unroll
    for (int t = 0; t < NTILES; ++t) {
        int col = l15 + (t << 4);
        if (TAIL && col >= outcols) continue;
        float bcol = bias ? bias[col] : 0.0f;
        #pragma unroll
        for (int r = 0; r < 8; ++r) {
            int row = row0 + r + (half << 3);
            float v = c[t][r];
            if (rowscale) v *= rowscale[row];
            out[(size_t)row * outld + col] = v + bcol;
        }
    }
}

// ---- edge aggregation: acc[dst] += hs[src], 64 f32 features ---------------
// One wave per edge; lane moves a float2. readfirstlane makes the edge id
// provably wave-uniform so src/dst come in via SMEM (s_load), keeping the
// vector memory path free for the gather + atomics (L2-resident).

__global__ void gcn_aggregate(const float* __restrict__ hs,
                              const long long* __restrict__ src,
                              const long long* __restrict__ dst,
                              float* __restrict__ acc, int e) {
    const int lane = threadIdx.x & 31;
    const int ed   = __builtin_amdgcn_readfirstlane(blockIdx.x * 8 + (threadIdx.x >> 5));
    if (ed >= e) return;
    const int s = (int)src[ed];
    const int d = (int)dst[ed];
    const float2 v = *(const float2*)(hs + (size_t)s * 64 + lane * 2);
    float* p = acc + (size_t)d * 64 + lane * 2;
    unsafeAtomicAdd(p + 0, v.x);
    unsafeAtomicAdd(p + 1, v.y);
}

// ---- combine: h = relu(dinv*(acc + hs) + b)  (self-loop = +hs) ------------

__global__ void gcn_combine(const float* __restrict__ acc, const float* __restrict__ hs,
                            const float* __restrict__ dinv, const float* __restrict__ bias,
                            float* __restrict__ h, int total) {
    int i = blockIdx.x * blockDim.x + threadIdx.x;
    if (i >= total) return;
    int row = i >> 6, f = i & 63;
    float v = dinv[row] * (acc[i] + hs[i]) + bias[f];
    h[i] = fmaxf(v, 0.0f);
}

// ---------------------------------------------------------------------------

extern "C" void kernel_launch(void* const* d_in, const int* in_sizes, int n_in,
                              void* d_out, int out_size, void* d_ws, size_t ws_size,
                              hipStream_t stream) {
    const float*     x   = (const float*)d_in[0];
    const long long* ei  = (const long long*)d_in[1];
    const float*     W1  = (const float*)d_in[2];
    const float*     b1  = (const float*)d_in[3];
    const float*     W2  = (const float*)d_in[4];
    const float*     b2  = (const float*)d_in[5];
    const float*     Wfc = (const float*)d_in[6];
    const float*     bfc = (const float*)d_in[7];
    float* out = (float*)d_out;

    const int N = in_sizes[0] / 128;   // 100000 (multiple of 16)
    const int E = in_sizes[1] / 2;     // 3200000
    const long long* src = ei;
    const long long* dst = ei + E;

    // workspace carve-out (all buffers fully rewritten each call)
    char* w = (char*)d_ws;
    auto carve = [&](size_t bytes) { char* p = w; w += (bytes + 255) & ~(size_t)255; return p; };
    float*          deg  = (float*)carve((size_t)N * 4);
    float*          dinv = (float*)carve((size_t)N * 4);
    float*          hs   = (float*)carve((size_t)N * 64 * 4);
    float*          acc  = (float*)carve((size_t)N * 64 * 4);
    float*          h    = (float*)carve((size_t)N * 64 * 4);
    unsigned short* wp1  = (unsigned short*)carve(4 * 4 * 512 * 2);  // K=128, N=64
    unsigned short* wp2  = (unsigned short*)carve(2 * 4 * 512 * 2);  // K=64,  N=64
    unsigned short* wpfc = (unsigned short*)carve(2 * 1 * 512 * 2);  // K=64,  N=11->16

    const int tiles      = N / 16;
    const int gemmBlocks = (tiles + 7) / 8;          // 8 waves (256 thr) per block
    const int nBlocks    = (N + 255) / 256;
    const int eBlocks    = (E + 255) / 256;
    const int aggBlocks  = (E + 7) / 8;
    const int cmbBlocks  = (N * 64 + 255) / 256;

    // weight packing + degrees -> dinv
    gcn_pack_w<<<(4 * 4 * 512 + 255) / 256, 256, 0, stream>>>(W1,  wp1,  4, 4, 64);
    gcn_pack_w<<<(2 * 4 * 512 + 255) / 256, 256, 0, stream>>>(W2,  wp2,  2, 4, 64);
    gcn_pack_w<<<(2 * 1 * 512 + 255) / 256, 256, 0, stream>>>(Wfc, wpfc, 2, 1, 11);
    gcn_init_deg<<<nBlocks, 256, 0, stream>>>(deg, N);
    gcn_degree  <<<eBlocks, 256, 0, stream>>>(dst, deg, E);
    gcn_dinv    <<<nBlocks, 256, 0, stream>>>(deg, dinv, N);

    // layer 1: hs = (x@W1)*dinv ; acc = scatter ; h = relu(dinv*(acc+hs)+b1)
    gcn_gemm<4, false><<<gemmBlocks, 256, 0, stream>>>(x, wp1, dinv, nullptr, hs, N, 128, 64, 64);
    hipMemsetAsync(acc, 0, (size_t)N * 64 * 4, stream);
    gcn_aggregate<<<aggBlocks, 256, 0, stream>>>(hs, src, dst, acc, E);
    gcn_combine  <<<cmbBlocks, 256, 0, stream>>>(acc, hs, dinv, b1, h, N * 64);

    // layer 2
    gcn_gemm<4, false><<<gemmBlocks, 256, 0, stream>>>(h, wp2, dinv, nullptr, hs, N, 64, 64, 64);
    hipMemsetAsync(acc, 0, (size_t)N * 64 * 4, stream);
    gcn_aggregate<<<aggBlocks, 256, 0, stream>>>(hs, src, dst, acc, E);
    gcn_combine  <<<cmbBlocks, 256, 0, stream>>>(acc, hs, dinv, b2, h, N * 64);

    // fc: out = h @ Wfc + bfc (packed B already zero-padded to 16 cols)
    gcn_gemm<1, true><<<gemmBlocks, 256, 0, stream>>>(h, wpfc, nullptr, bfc, out, N, 64, 11, 11);
}
